// LLaMA3ARModel_25580825215371
// MI455X (gfx1250) — compile-verified, matrix operand order checked by reading, not run on hardware
//
#include <hip/hip_runtime.h>
#include <hip/hip_bf16.h>
#include <stdint.h>

// ---------------------------------------------------------------------------
// Types for CDNA5 WMMA (gfx1250, wave32)
// ---------------------------------------------------------------------------
typedef __attribute__((ext_vector_type(16))) __bf16 v16bf;
typedef __attribute__((ext_vector_type(8)))  float  v8f;
typedef __attribute__((ext_vector_type(4)))  unsigned int v4u;
typedef unsigned short u16b;

union FragAB {
    v16bf v;
    v4u   q[2];
    u16b  u[16];
};

#define BATCH   2
#define SEQ     2048
#define TOK     4096        // BATCH*SEQ
#define DIM     1024
#define NLAYER  8
#define NH      16
#define NKVH    8
#define HDIM    64
#define FFD     4096
#define VOCAB   4054
#define PAD_ID  4053
#define RMSEPS  1e-5f
#define NEGINF  (-1e30f)

__device__ __forceinline__ u16b f2bf(float f) {
    union { float f; unsigned u; } x; x.f = f;
    unsigned r = (x.u + 0x7FFFu + ((x.u >> 16) & 1u)) >> 16;
    return (u16b)r;
}

// CDNA5 async global->LDS copy (16B per lane), tracked by ASYNCcnt.
// ldsoff = wave-relative LDS byte address (low 32 bits of generic pointer),
// gaddr  = global address (VGPR pair).
__device__ __forceinline__ void async_copy_b128(unsigned ldsoff, const void* gaddr) {
    asm volatile("global_load_async_to_lds_b128 %0, %1, off"
                 :: "v"(ldsoff), "v"(gaddr) : "memory");
}
__device__ __forceinline__ void async_wait0() {
    asm volatile("s_wait_asynccnt 0x0" ::: "memory");
}

// ---------------------------------------------------------------------------
// Embedding gather: x[t][d] = emb[ids[t]][d]
// ---------------------------------------------------------------------------
__global__ void k_embed(const int* __restrict__ ids, const float* __restrict__ emb,
                        float* __restrict__ x) {
    int i = blockIdx.x * 256 + threadIdx.x;         // < TOK*DIM
    int t = i >> 10, d = i & 1023;
    x[i] = emb[(size_t)ids[t] * DIM + d];
}

// Key bias: -inf for pad keys (applies to all layers, computed once)
__global__ void k_kbias(const int* __restrict__ ids, float* __restrict__ kbias) {
    int i = blockIdx.x * 256 + threadIdx.x;
    if (i < BATCH * SEQ) kbias[i] = (ids[i] == PAD_ID) ? NEGINF : 0.f;
}

// ---------------------------------------------------------------------------
// RMSNorm (fp32 in) -> bf16 out.  One block (256 threads) per token.
// ---------------------------------------------------------------------------
__global__ void k_rmsnorm(const float* __restrict__ x, const float* __restrict__ w,
                          u16b* __restrict__ out) {
    int t = blockIdx.x;
    const float* xr = x + (size_t)t * DIM;
    u16b* orow = out + (size_t)t * DIM;
    __shared__ float red[256];
    float s = 0.f;
    for (int d = threadIdx.x; d < DIM; d += 256) { float v = xr[d]; s += v * v; }
    red[threadIdx.x] = s;
    __syncthreads();
    for (int st = 128; st > 0; st >>= 1) {
        if (threadIdx.x < st) red[threadIdx.x] += red[threadIdx.x + st];
        __syncthreads();
    }
    float rs = rsqrtf(red[0] / (float)DIM + RMSEPS);
    for (int d = threadIdx.x; d < DIM; d += 256) orow[d] = f2bf(xr[d] * rs * w[d]);
}

// ---------------------------------------------------------------------------
// WMMA GEMM: C[M,N](fp32) (+)= A[M,K](bf16,row-major) x B[K,N](fp32,row-major)
//
// Block: 256 threads = 8 waves in a 4x2 grid. Block tile 128(M) x 128(N),
// k-step 32. Each wave owns 32x64 (2 A-frags x 4 B-frags -> 8 WMMAs/step).
//  - A tile (128x32 bf16, 8 KB) is staged raw via GLOBAL_LOAD_ASYNC_TO_LDS_B128
//    (ASYNCcnt path; no VGPR round trip, overlaps with B conversion).
//  - B tile (32x128 fp32) is read coalesced, converted fp32->bf16, stored
//    transposed [n][k] so B-frags are contiguous ds_load_b128.
// EPI: 0 = store, 1 = accumulate (residual add into C).
// ---------------------------------------------------------------------------
template<int EPI>
__global__ void k_gemm_bf16(const u16b* __restrict__ A, const float* __restrict__ B,
                            float* __restrict__ C, int M, int N, int K) {
    const int tid = threadIdx.x;
    const int wave = tid >> 5, lane = tid & 31;
    const int laneLo = lane & 15, hiHalf = lane >> 4;
    const int wr = wave >> 1, wc = wave & 1;           // 4x2 wave grid
    const int m0 = blockIdx.x * 128;
    const int nb = blockIdx.y * 128;
    const int koff = hiHalf ? 8 : 0;

    __shared__ u16b ldsA[128 * 32];                    // 8 KB, [m][k]
    __shared__ u16b ldsB[128 * 32];                    // 8 KB, transposed [n][k]

    v8f acc[2][4] = { { {}, {}, {}, {} }, { {}, {}, {}, {} } };

    for (int k0 = 0; k0 < K; k0 += 32) {
        // ---- A tile: async raw copy, 512 x 16B chunks, 2 per thread ----
        for (int c = tid; c < 512; c += 256) {
            int row = c >> 2, seg = c & 3;             // 4 x 16B per 64B row
            unsigned loff = (unsigned)(size_t)&ldsA[row * 32 + seg * 8];
            const u16b* gp = A + (size_t)(m0 + row) * K + k0 + seg * 8;
            async_copy_b128(loff, gp);
        }
        // ---- B tile: coalesced fp32 read + cvt, store transposed ----
        for (int i = tid; i < 32 * 128; i += 256) {
            int kk = i >> 7, nn = i & 127;
            int gn = nb + nn;
            float bv = (gn < N) ? B[(size_t)(k0 + kk) * N + gn] : 0.f;
            ldsB[nn * 32 + kk] = f2bf(bv);
        }
        async_wait0();
        __syncthreads();

        FragAB af[2], bfr[4];
#pragma unroll
        for (int r = 0; r < 2; r++) {
            const u16b* ap = &ldsA[(wr * 32 + r * 16 + laneLo) * 32 + koff];
            af[r].q[0] = *(const v4u*)ap;
            af[r].q[1] = *(const v4u*)(ap + 16);
        }
#pragma unroll
        for (int j = 0; j < 4; j++) {
            const u16b* bp = &ldsB[(wc * 64 + j * 16 + laneLo) * 32 + koff];
            bfr[j].q[0] = *(const v4u*)bp;
            bfr[j].q[1] = *(const v4u*)(bp + 16);
        }
#pragma unroll
        for (int r = 0; r < 2; r++)
#pragma unroll
            for (int j = 0; j < 4; j++)
                acc[r][j] = __builtin_amdgcn_wmma_f32_16x16x32_bf16(
                    false, af[r].v, false, bfr[j].v, (short)0, acc[r][j], false, false);
        __syncthreads();
    }

#pragma unroll
    for (int r = 0; r < 2; r++) {
#pragma unroll
        for (int j = 0; j < 4; j++) {
            int col = nb + wc * 64 + j * 16 + laneLo;
            if (col < N) {
#pragma unroll
                for (int e = 0; e < 8; e++) {
                    int row = m0 + wr * 32 + r * 16 + e + hiHalf * 8;
                    size_t off = (size_t)row * N + col;
                    if (EPI == 0) C[off] = acc[r][j][e];
                    else          C[off] += acc[r][j][e];
                }
            }
        }
    }
}

// ---------------------------------------------------------------------------
// RoPE: fp32 projection [TOK, nh*64] -> bf16 [B, nh, S, 64]
// ---------------------------------------------------------------------------
__global__ void k_rope(const float* __restrict__ src, u16b* __restrict__ dst, int nheads) {
    int i = blockIdx.x * 256 + threadIdx.x;       // < TOK*nheads*64
    int cols = nheads * HDIM;
    int t = i / cols;
    int rem = i - t * cols;
    int h = rem >> 6, d = rem & 63;
    int s = t & (SEQ - 1), b = t >> 11;
    int io = d & 31;
    float freq = __expf(-(2.0f * (float)io / (float)HDIM) * __logf(500000.0f));
    float ang = (float)s * freq;
    const float* row = src + (size_t)t * cols;
    float v = row[rem];
    float o = (d < 32) ? -row[rem + 32] : row[rem - 32];
    float out = v * __cosf(ang) + o * __sinf(ang);
    dst[(((size_t)b * nheads + h) * SEQ + s) * HDIM + d] = f2bf(out);
}

// V: fp32 [TOK, KVH*64] -> bf16 transposed [B, KVH, 64, S] (contiguous in kv)
__global__ void k_vtrans(const float* __restrict__ src, u16b* __restrict__ dst) {
    int i = blockIdx.x * 256 + threadIdx.x;       // < TOK*512
    int t = i >> 9, rem = i & 511;
    int kvh = rem >> 6, d = rem & 63;
    int s = t & (SEQ - 1), b = t >> 11;
    dst[(((size_t)b * NKVH + kvh) * HDIM + d) * SEQ + s] = f2bf(src[i]);
}

// SwiGLU: h = silu(g) * u  -> bf16
__global__ void k_silumul(const float* __restrict__ g, const float* __restrict__ u,
                          u16b* __restrict__ h) {
    int i = blockIdx.x * 256 + threadIdx.x;
    float gv = g[i];
    float sg = gv / (1.f + __expf(-gv));
    h[i] = f2bf(sg * u[i]);
}

// ---------------------------------------------------------------------------
// Flash attention (causal + pad-key bias), GQA (NREP=2), WMMA for QK^T and PV.
// Block: 128 threads = 4 waves; each wave owns a 16-row Q tile, 64 q rows/block.
// grid = (SEQ/64, B*H).
// ---------------------------------------------------------------------------
__global__ void k_attention(const u16b* __restrict__ qb, const u16b* __restrict__ kb,
                            const u16b* __restrict__ vt, const float* __restrict__ kbias,
                            u16b* __restrict__ ob) {
    const int tid = threadIdx.x;
    const int wave = tid >> 5, lane = tid & 31;
    const int laneLo = lane & 15, hiHalf = lane >> 4;
    const int bh = blockIdx.y;
    const int b = bh / NH, h = bh % NH;
    const int kvh = h >> 1;
    const int qbase = blockIdx.x * 64 + wave * 16;
    const float scale = 0.125f;                    // 1/sqrt(64)

    const u16b* qptr = qb + (((size_t)b * NH + h) * SEQ) * HDIM;
    const u16b* kptr = kb + (((size_t)b * NKVH + kvh) * SEQ) * HDIM;
    const u16b* vptr = vt + (((size_t)b * NKVH + kvh) * HDIM) * SEQ;
    const float* biasb = kbias + (size_t)b * SEQ;

    __shared__ u16b plds[4][16 * 32];              // per-wave P tile (bf16)

    // Q fragments, contraction HD=64 -> two k-chunks of 32
    FragAB qa0, qa1;
    {
        const u16b* qp = qptr + (size_t)(qbase + laneLo) * HDIM + (hiHalf ? 8 : 0);
        qa0.q[0] = *(const v4u*)qp;        qa0.q[1] = *(const v4u*)(qp + 16);
        qa1.q[0] = *(const v4u*)(qp + 32); qa1.q[1] = *(const v4u*)(qp + 48);
    }

    float mrun[8], lrun[8];
#pragma unroll
    for (int e = 0; e < 8; e++) { mrun[e] = NEGINF; lrun[e] = 0.f; }
    v8f accO[4] = { {}, {}, {}, {} };

    const int qhi = qbase + 15;
    for (int c = 0; c <= qhi; c += 32) {
        // ----- scores: two 16-col kv tiles -----
        v8f sc[2];
#pragma unroll
        for (int tnum = 0; tnum < 2; tnum++) {
            int kvj = c + tnum * 16 + laneLo;
            FragAB kf0, kf1;
            if (kvj < SEQ) {
                const u16b* kp = kptr + (size_t)kvj * HDIM + (hiHalf ? 8 : 0);
                kf0.q[0] = *(const v4u*)kp;        kf0.q[1] = *(const v4u*)(kp + 16);
                kf1.q[0] = *(const v4u*)(kp + 32); kf1.q[1] = *(const v4u*)(kp + 48);
            } else {
                v4u z = {0u, 0u, 0u, 0u};
                kf0.q[0] = z; kf0.q[1] = z; kf1.q[0] = z; kf1.q[1] = z;
            }
            v8f s = {};
            s = __builtin_amdgcn_wmma_f32_16x16x32_bf16(false, qa0.v, false, kf0.v, (short)0, s, false, false);
            s = __builtin_amdgcn_wmma_f32_16x16x32_bf16(false, qa1.v, false, kf1.v, (short)0, s, false, false);
            sc[tnum] = s;
        }

        int j0 = c + laneLo, j1 = c + 16 + laneLo;
        float bias0 = (j0 < SEQ) ? biasb[j0] : 0.f;
        float bias1 = (j1 < SEQ) ? biasb[j1] : 0.f;

        float a0[8], a1[8], rm[8];
#pragma unroll
        for (int e = 0; e < 8; e++) {
            int qi = qbase + e + hiHalf * 8;
            float x0 = sc[0][e] * scale + bias0; if (j0 > qi || j0 >= SEQ) x0 = NEGINF;
            float x1 = sc[1][e] * scale + bias1; if (j1 > qi || j1 >= SEQ) x1 = NEGINF;
            a0[e] = x0; a1[e] = x1;
            float r = fmaxf(x0, x1);
            r = fmaxf(r, __shfl_xor(r, 1, 32));
            r = fmaxf(r, __shfl_xor(r, 2, 32));
            r = fmaxf(r, __shfl_xor(r, 4, 32));
            r = fmaxf(r, __shfl_xor(r, 8, 32));
            rm[e] = r;
        }

        // online softmax update + stash P (bf16) in per-wave LDS
#pragma unroll
        for (int e = 0; e < 8; e++) {
            float nm = fmaxf(mrun[e], rm[e]);
            float sf = __expf(mrun[e] - nm);
            mrun[e] = nm;
            float e0 = __expf(a0[e] - nm);
            float e1 = __expf(a1[e] - nm);
            float rs = e0 + e1;
            rs += __shfl_xor(rs, 1, 32);
            rs += __shfl_xor(rs, 2, 32);
            rs += __shfl_xor(rs, 4, 32);
            rs += __shfl_xor(rs, 8, 32);
            lrun[e] = lrun[e] * sf + rs;
#pragma unroll
            for (int j = 0; j < 4; j++) accO[j][e] *= sf;
            int prow = e + hiHalf * 8;
            plds[wave][prow * 32 + laneLo]      = f2bf(e0);
            plds[wave][prow * 32 + 16 + laneLo] = f2bf(e1);
        }

        // P fragment (16x32, contraction = kv)
        FragAB pa;
        {
            const u16b* pp = &plds[wave][laneLo * 32 + (hiHalf ? 8 : 0)];
            pa.q[0] = *(const v4u*)pp;
            pa.q[1] = *(const v4u*)(pp + 16);
        }

        // V fragments (from transposed V, contiguous in kv) + accumulate
#pragma unroll
        for (int j = 0; j < 4; j++) {
            int hd = j * 16 + laneLo;
            const u16b* vp = vptr + (size_t)hd * SEQ + c + (hiHalf ? 8 : 0);
            FragAB vfr;
            if (c + 32 <= SEQ) {
                vfr.q[0] = *(const v4u*)vp;
                vfr.q[1] = *(const v4u*)(vp + 16);
            } else {
#pragma unroll
                for (int ii = 0; ii < 8; ii++) {
                    int kv0 = c + (hiHalf ? 8 : 0) + ii;
                    int kv1 = kv0 + 16;
                    vfr.u[ii]     = (kv0 < SEQ) ? vp[ii]      : (u16b)0;
                    vfr.u[ii + 8] = (kv1 < SEQ) ? vp[ii + 16] : (u16b)0;
                }
            }
            accO[j] = __builtin_amdgcn_wmma_f32_16x16x32_bf16(
                false, pa.v, false, vfr.v, (short)0, accO[j], false, false);
        }
    }

    // normalize and store O (bf16, [TOK, H*HD])
#pragma unroll
    for (int e = 0; e < 8; e++) {
        float inv = 1.0f / lrun[e];
        int srow = qbase + e + hiHalf * 8;
        size_t tix = (size_t)b * SEQ + srow;
#pragma unroll
        for (int j = 0; j < 4; j++) {
            ob[tix * (NH * HDIM) + h * HDIM + j * 16 + laneLo] = f2bf(accO[j][e] * inv);
        }
    }
}

// ---------------------------------------------------------------------------
// Host orchestration
// ---------------------------------------------------------------------------
extern "C" void kernel_launch(void* const* d_in, const int* in_sizes, int n_in,
                              void* d_out, int out_size, void* d_ws, size_t ws_size,
                              hipStream_t stream) {
    (void)in_sizes; (void)n_in; (void)out_size; (void)ws_size;
    const int*   ids  = (const int*)d_in[0];
    const float* emb  = (const float*)d_in[1];
    const float* Wq   = (const float*)d_in[2];
    const float* Wk   = (const float*)d_in[3];
    const float* Wv   = (const float*)d_in[4];
    const float* Wo   = (const float*)d_in[5];
    const float* Wg   = (const float*)d_in[6];
    const float* Wu   = (const float*)d_in[7];
    const float* Wd   = (const float*)d_in[8];
    const float* anw  = (const float*)d_in[9];
    const float* fnw  = (const float*)d_in[10];
    const float* finw = (const float*)d_in[11];
    const float* lmh  = (const float*)d_in[12];
    float* out = (float*)d_out;

    char* ws = (char*)d_ws;
    size_t off = 0;
    auto alloc = [&](size_t bytes) -> void* {
        void* p = ws + off;
        off = (off + bytes + 255) & ~(size_t)255;
        return p;
    };
    float* x     = (float*)alloc((size_t)TOK * DIM * 4);
    u16b*  hn    = (u16b*) alloc((size_t)TOK * DIM * 2);
    float* qf    = (float*)alloc((size_t)TOK * 1024 * 4);
    float* kf    = (float*)alloc((size_t)TOK * 512 * 4);
    float* vf    = (float*)alloc((size_t)TOK * 512 * 4);
    u16b*  qbb   = (u16b*) alloc((size_t)TOK * 1024 * 2);
    u16b*  kbb   = (u16b*) alloc((size_t)TOK * 512 * 2);
    u16b*  vtt   = (u16b*) alloc((size_t)TOK * 512 * 2);
    u16b*  obb   = (u16b*) alloc((size_t)TOK * 1024 * 2);
    float* gb    = (float*)alloc((size_t)TOK * FFD * 4);
    float* ub    = (float*)alloc((size_t)TOK * FFD * 4);
    u16b*  hb    = (u16b*) alloc((size_t)TOK * FFD * 2);
    float* kbias = (float*)alloc((size_t)BATCH * SEQ * 4);

    k_embed<<<(TOK * DIM) / 256, 256, 0, stream>>>(ids, emb, x);
    k_kbias<<<(BATCH * SEQ) / 256, 256, 0, stream>>>(ids, kbias);

    dim3 gQ (TOK / 128, (1024 + 127) / 128);   // (32, 8)
    dim3 gKV(TOK / 128, (512 + 127) / 128);    // (32, 4)
    dim3 gFF(TOK / 128, FFD / 128);            // (32, 32)
    dim3 gLM(TOK / 128, (VOCAB + 127) / 128);  // (32, 32), tail guarded

    for (int l = 0; l < NLAYER; l++) {
        k_rmsnorm<<<TOK, 256, 0, stream>>>(x, anw + (size_t)l * DIM, hn);
        k_gemm_bf16<0><<<gQ,  256, 0, stream>>>(hn, Wq + (size_t)l * DIM * 1024, qf, TOK, 1024, DIM);
        k_gemm_bf16<0><<<gKV, 256, 0, stream>>>(hn, Wk + (size_t)l * DIM * 512,  kf, TOK, 512,  DIM);
        k_gemm_bf16<0><<<gKV, 256, 0, stream>>>(hn, Wv + (size_t)l * DIM * 512,  vf, TOK, 512,  DIM);
        k_rope<<<(TOK * 1024) / 256, 256, 0, stream>>>(qf, qbb, NH);
        k_rope<<<(TOK * 512)  / 256, 256, 0, stream>>>(kf, kbb, NKVH);
        k_vtrans<<<(TOK * 512) / 256, 256, 0, stream>>>(vf, vtt);
        k_attention<<<dim3(SEQ / 64, BATCH * NH), 128, 0, stream>>>(qbb, kbb, vtt, kbias, obb);
        k_gemm_bf16<1><<<gQ, 256, 0, stream>>>(obb, Wo + (size_t)l * 1024 * DIM, x, TOK, DIM, 1024);
        k_rmsnorm<<<TOK, 256, 0, stream>>>(x, fnw + (size_t)l * DIM, hn);
        k_gemm_bf16<0><<<gFF, 256, 0, stream>>>(hn, Wg + (size_t)l * DIM * FFD, gb, TOK, FFD, DIM);
        k_gemm_bf16<0><<<gFF, 256, 0, stream>>>(hn, Wu + (size_t)l * DIM * FFD, ub, TOK, FFD, DIM);
        k_silumul<<<(TOK * FFD) / 256, 256, 0, stream>>>(gb, ub, hb);
        k_gemm_bf16<1><<<gQ, 256, 0, stream>>>(hb, Wd + (size_t)l * FFD * DIM, x, TOK, DIM, FFD);
    }

    k_rmsnorm<<<TOK, 256, 0, stream>>>(x, finw, hn);
    k_gemm_bf16<0><<<gLM, 256, 0, stream>>>(hn, lmh, out, TOK, VOCAB, DIM);
}
